// DrugNet_15676630631062
// MI455X (gfx1250) — matrix operand order repaired and checked by reading, not compiled, not run on hardware
//
#include <hip/hip_runtime.h>
#include <cstdint>
#include <cstddef>

// ======================= CDNA5 (gfx1250) DrugNet forward =======================
// Dense linear algebra runs through v_wmma_f32_16x16x32_f16 (f16 in, f32 acc).
// Irregular graph ops (segment softmax, scatter) use f32 atomics.

#define DEVFN __device__ __forceinline__
#define CEILDIV(a,b) (((a)+(b)-1)/(b))

typedef _Float16 v16h __attribute__((ext_vector_type(16)));
typedef _Float16 v8h  __attribute__((ext_vector_type(8)));
typedef __fp16   v2hf __attribute__((ext_vector_type(2)));   // cvt_pkrtz result type
typedef float    v8f  __attribute__((ext_vector_type(8)));

enum { ACT_NONE = 0, ACT_RELU = 1, ACT_LEAKY = 2, ACT_ELU = 3 };

DEVFN float act_apply(float v, int act) {
  if (act == ACT_RELU)  return v > 0.f ? v : 0.f;
  if (act == ACT_LEAKY) return v >= 0.f ? v : 0.01f * v;
  if (act == ACT_ELU)   return v > 0.f ? v : (__expf(v) - 1.f);
  return v;
}
DEVFN float sigmoidf(float x) { return 1.f / (1.f + __expf(-x)); }
DEVFN float leakyf(float x)   { return x >= 0.f ? x : 0.01f * x; }

// monotonic float <-> uint mapping for atomicMax-based segment max
DEVFN unsigned f2ord(float f) {
  unsigned u = __float_as_uint(f);
  return (u & 0x80000000u) ? ~u : (u | 0x80000000u);
}
DEVFN float ord2f(unsigned u) {
  u = (u & 0x80000000u) ? (u & 0x7fffffffu) : ~u;
  return __uint_as_float(u);
}

// pack 8 f32 -> 8 f16 via v_cvt_pk_rtz_f16_f32, with 0/1 validity scale
DEVFN v8h cvt8(float4 x, float4 y, float s) {
  v2hf p0 = __builtin_amdgcn_cvt_pkrtz(x.x * s, x.y * s);
  v2hf p1 = __builtin_amdgcn_cvt_pkrtz(x.z * s, x.w * s);
  v2hf p2 = __builtin_amdgcn_cvt_pkrtz(y.x * s, y.y * s);
  v2hf p3 = __builtin_amdgcn_cvt_pkrtz(y.z * s, y.w * s);
  v8h r;
  r[0] = (_Float16)p0[0]; r[1] = (_Float16)p0[1];
  r[2] = (_Float16)p1[0]; r[3] = (_Float16)p1[1];
  r[4] = (_Float16)p2[0]; r[5] = (_Float16)p2[1];
  r[6] = (_Float16)p3[0]; r[7] = (_Float16)p3[1];
  return r;
}

// branchless 8-wide f32 chunk prefetch with clamped (always-legal) addressing.
// Requires K % 8 == 0 (true for every GEMM in this model), so a chunk is
// either fully valid or fully invalid -> single 0/1 scale, no divergence.
DEVFN void pf8(const float* __restrict__ base, int row, int ld, int kb,
               int rmax, int K, float4& x, float4& y, float& s) {
  const int rc = row < rmax ? row : (rmax - 1);
  const int kc = kb <= (K - 8) ? kb : (K - 8);
  const float4* p = (const float4*)(base + (size_t)rc * ld + kc);
  x = p[0];
  y = p[1];
  s = (row < rmax && kb < K) ? 1.f : 0.f;
}

// LDS fragment loads (layouts per CDNA5 ISA 7.12.2, wave32)
constexpr int LSTR = 40; // LDS row stride in halves (80B, 16B-aligned)

DEVFN v16h ld_afrag(const _Float16* rowp, int half) {
  // A 16x32 f16: lanes 0-15 K 0..7/16..23 ; lanes 16-31 K 8..15/24..31
  v8h lo = *reinterpret_cast<const v8h*>(rowp + half * 8);
  v8h hi = *reinterpret_cast<const v8h*>(rowp + 16 + half * 8);
  v16h r;
#pragma unroll
  for (int i = 0; i < 8; ++i) { r[i] = lo[i]; r[8 + i] = hi[i]; }
  return r;
}
DEVFN v16h ld_bfrag(const _Float16* rowp, int half) {
  // B 32x16 f16: lane holds col N=l16; lanes 0-15 K 0..7/16..23 ; lanes 16-31 K 8..15/24..31
  v8h lo = *reinterpret_cast<const v8h*>(rowp + half * 8);
  v8h hi = *reinterpret_cast<const v8h*>(rowp + 16 + half * 8);
  v16h r;
#pragma unroll
  for (int i = 0; i < 8; ++i) { r[i] = lo[i]; r[8 + i] = hi[i]; }
  return r;
}

// ---------------------------------------------------------------------------
// WMMA GEMM: C[M,Ncols] = act(A[M,K] @ W[Ncols,K]^T + bias)
// Block tile 128(M) x 64(N), 256 threads = 8 waves, each wave owns a 32x32
// sub-tile (4 accumulators -> 4 WMMA per K-chunk from 8 ds_load_b128).
// Double-buffered LDS with register prefetch of the next K-chunk.
// ---------------------------------------------------------------------------
__global__ __launch_bounds__(256) void dn_gemm_wmma(
    const float* __restrict__ A, int lda,
    const float* __restrict__ W, int ldw,
    const float* __restrict__ bias,
    float* __restrict__ C, int ldc,
    int M, int Ncols, int K, int act)
{
  __shared__ __align__(16) _Float16 As[2 * 128 * LSTR];
  __shared__ __align__(16) _Float16 Bs[2 * 64 * LSTR];

  const int tid  = threadIdx.x;
  const int lane = tid & 31;
  const int wave = tid >> 5;
  const int wm   = wave >> 1;             // 0..3 : 32-row M sub-tile
  const int wn   = wave & 1;              // 0..1 : 32-col N sub-tile
  const int half = lane >> 4;             // 0/1
  const int l16  = lane & 15;

  const int bm = blockIdx.y * 128;
  const int bn = blockIdx.x * 64;

  v8f acc00 = {}, acc01 = {}, acc10 = {}, acc11 = {};

  // cooperative staging: A = 128x32 (2 chunks/thread), B = 64x32 (1 chunk)
  const int ar0 = tid >> 2;               // 0..63
  const int ar1 = ar0 + 64;               // 64..127
  const int ldk = (tid & 3) * 8;          // 0,8,16,24

  float4 ax0, ay0, ax1, ay1, bx, by;
  float  sa0, sa1, sb;
  pf8(A, bm + ar0, lda, ldk, M, K, ax0, ay0, sa0);
  pf8(A, bm + ar1, lda, ldk, M, K, ax1, ay1, sa1);
  pf8(W, bn + ar0, ldw, ldk, Ncols, K, bx, by, sb);

  int buf = 0;
  for (int k0 = 0; k0 < K; k0 += 32) {
    _Float16* Asb = As + buf * (128 * LSTR);
    _Float16* Bsb = Bs + buf * (64 * LSTR);
    *reinterpret_cast<v8h*>(&Asb[ar0 * LSTR + ldk]) = cvt8(ax0, ay0, sa0);
    *reinterpret_cast<v8h*>(&Asb[ar1 * LSTR + ldk]) = cvt8(ax1, ay1, sa1);
    *reinterpret_cast<v8h*>(&Bsb[ar0 * LSTR + ldk]) = cvt8(bx, by, sb);

    if (k0 + 32 < K) { // uniform scalar branch: prefetch next chunk
      pf8(A, bm + ar0, lda, k0 + 32 + ldk, M, K, ax0, ay0, sa0);
      pf8(A, bm + ar1, lda, k0 + 32 + ldk, M, K, ax1, ay1, sa1);
      pf8(W, bn + ar0, ldw, k0 + 32 + ldk, Ncols, K, bx, by, sb);
    }
    __syncthreads();

    const v16h af0 = ld_afrag(&Asb[(wm * 32 + l16) * LSTR], half);
    const v16h af1 = ld_afrag(&Asb[(wm * 32 + 16 + l16) * LSTR], half);
    const v16h bf0 = ld_bfrag(&Bsb[(wn * 32 + l16) * LSTR], half);
    const v16h bf1 = ld_bfrag(&Bsb[(wn * 32 + 16 + l16) * LSTR], half);

    acc00 = __builtin_amdgcn_wmma_f32_16x16x32_f16(false, af0, false, bf0, (short)0, acc00, false, false);
    acc01 = __builtin_amdgcn_wmma_f32_16x16x32_f16(false, af0, false, bf1, (short)0, acc01, false, false);
    acc10 = __builtin_amdgcn_wmma_f32_16x16x32_f16(false, af1, false, bf0, (short)0, acc10, false, false);
    acc11 = __builtin_amdgcn_wmma_f32_16x16x32_f16(false, af1, false, bf1, (short)0, acc11, false, false);

    buf ^= 1;
  }

  // Epilogue. C/D layout: lane = col (mod 16), VGPR i -> row = half*8 + i.
#pragma unroll
  for (int tm = 0; tm < 2; ++tm) {
#pragma unroll
    for (int tn = 0; tn < 2; ++tn) {
      const v8f acc = tm ? (tn ? acc11 : acc10) : (tn ? acc01 : acc00);
      const int col = bn + wn * 32 + tn * 16 + l16;
      if (col < Ncols) {
        const float bv = bias ? bias[col] : 0.f;
#pragma unroll
        for (int i = 0; i < 8; ++i) {
          const int row = bm + wm * 32 + tm * 16 + half * 8 + i;
          if (row < M) {
            C[(size_t)row * ldc + col] = act_apply(acc[i] + bv, act);
          }
        }
      }
    }
  }
}

// ------------------------------ small kernels ------------------------------
__global__ void dn_fill(float* __restrict__ p, float v, int n) {
  int i = blockIdx.x * blockDim.x + threadIdx.x;
  if (i < n) p[i] = v;
}

__global__ void dn_rownorm16(const float* __restrict__ ea, float* __restrict__ out, int E) {
  int e = blockIdx.x * blockDim.x + threadIdx.x;
  if (e >= E) return;
  const float* r = ea + (size_t)e * 16;
  float s = 0.f;
#pragma unroll
  for (int j = 0; j < 16; ++j) s += r[j] * r[j];
  float inv = 1.f / fmaxf(__fsqrt_rn(s), 1e-12f);
  float* o = out + (size_t)e * 16;
#pragma unroll
  for (int j = 0; j < 16; ++j) o[j] = r[j] * inv;
}

// z[e,:] = leaky(xs1[src[e],:] + z[e,:])   (E*128 threads)
__global__ void dn_edge_gate_z(const float* __restrict__ xs1, float* __restrict__ z,
                               const int* __restrict__ src, int total) {
  int i = blockIdx.x * blockDim.x + threadIdx.x;
  if (i >= total) return;
  int e = i >> 7, j = i & 127;
  z[i] = leakyf(xs1[(size_t)src[e] * 128 + j] + z[i]);
}

// out[r] = dot(X[r,0:128], v)
__global__ void dn_dot(const float* __restrict__ X, const float* __restrict__ v,
                       float* __restrict__ out, int rows) {
  int r = blockIdx.x * blockDim.x + threadIdx.x;
  if (r >= rows) return;
  const float4* x4 = (const float4*)(X + (size_t)r * 128);
  const float4* v4 = (const float4*)v;
  float s = 0.f;
#pragma unroll
  for (int i = 0; i < 32; ++i) {
    float4 a = x4[i], b = v4[i];
    s += a.x * b.x + a.y * b.y + a.z * b.z + a.w * b.w;
  }
  out[r] = s;
}

__global__ void dn_dot2(const float* __restrict__ X, const float* __restrict__ v1,
                        const float* __restrict__ v2, float* __restrict__ o1,
                        float* __restrict__ o2, int rows) {
  int r = blockIdx.x * blockDim.x + threadIdx.x;
  if (r >= rows) return;
  const float4* x4 = (const float4*)(X + (size_t)r * 128);
  const float4* a4 = (const float4*)v1;
  const float4* b4 = (const float4*)v2;
  float s1 = 0.f, s2 = 0.f;
#pragma unroll
  for (int i = 0; i < 32; ++i) {
    float4 x = x4[i], a = a4[i], b = b4[i];
    s1 += x.x * a.x + x.y * a.y + x.z * a.z + x.w * a.w;
    s2 += x.x * b.x + x.y * b.y + x.z * b.z + x.w * b.w;
  }
  o1[r] = s1; o2[r] = s2;
}

// a[e] = leaky(dot(z[e], att_l) + xr[dst[e]])
__global__ void dn_edge_gate_score(const float* __restrict__ z, const float* __restrict__ attl,
                                   const float* __restrict__ xr, const int* __restrict__ dst,
                                   float* __restrict__ a, int E) {
  int e = blockIdx.x * blockDim.x + threadIdx.x;
  if (e >= E) return;
  const float4* z4 = (const float4*)(z + (size_t)e * 128);
  const float4* a4 = (const float4*)attl;
  float s = 0.f;
#pragma unroll
  for (int i = 0; i < 32; ++i) {
    float4 x = z4[i], w = a4[i];
    s += x.x * w.x + x.y * w.y + x.z * w.z + x.w * w.w;
  }
  a[e] = leakyf(s + xr[dst[e]]);
}

__global__ void dn_edge_score(const float* __restrict__ asn, const float* __restrict__ adn,
                              const int* __restrict__ src, const int* __restrict__ dst,
                              float* __restrict__ a, int E) {
  int e = blockIdx.x * blockDim.x + threadIdx.x;
  if (e >= E) return;
  a[e] = leakyf(asn[src[e]] + adn[dst[e]]);
}

__global__ void dn_mol_score(const float* __restrict__ asn, const float* __restrict__ adg,
                             const int* __restrict__ batch, float* __restrict__ a, int N) {
  int n = blockIdx.x * blockDim.x + threadIdx.x;
  if (n >= N) return;
  a[n] = leakyf(asn[n] + adg[batch[n]]);
}

__global__ void dn_seg_max(const float* __restrict__ a, const int* __restrict__ idx,
                           unsigned* __restrict__ mu, int n) {
  int i = blockIdx.x * blockDim.x + threadIdx.x;
  if (i >= n) return;
  atomicMax(&mu[idx[i]], f2ord(a[i]));
}

__global__ void dn_seg_exp(const float* __restrict__ a, const int* __restrict__ idx,
                           const unsigned* __restrict__ mu, float* __restrict__ e, int n) {
  int i = blockIdx.x * blockDim.x + threadIdx.x;
  if (i >= n) return;
  unsigned u = mu[idx[i]];
  float m = (u == 0u) ? 0.f : ord2f(u);   // untouched sentinel -> 0 (matches isfinite branch)
  e[i] = __expf(a[i] - m);
}

__global__ void dn_seg_sum(const float* __restrict__ e, const int* __restrict__ idx,
                           float* __restrict__ s, int n) {
  int i = blockIdx.x * blockDim.x + threadIdx.x;
  if (i >= n) return;
  atomicAdd(&s[idx[i]], e[i]);
}

__global__ void dn_seg_alpha(const float* __restrict__ e, const int* __restrict__ idx,
                             const float* __restrict__ s, float* __restrict__ al, int n) {
  int i = blockIdx.x * blockDim.x + threadIdx.x;
  if (i >= n) return;
  al[i] = e[i] / (s[idx[i]] + 1e-16f);
}

// out[dst[e],:] += al[e] * Z[e,:]
__global__ void dn_scatter_edge_feat(const float* __restrict__ Z, const float* __restrict__ al,
                                     const int* __restrict__ dst, float* __restrict__ out, int total) {
  int i = blockIdx.x * blockDim.x + threadIdx.x;
  if (i >= total) return;
  int e = i >> 7, j = i & 127;
  atomicAdd(&out[(size_t)dst[e] * 128 + j], al[e] * Z[i]);
}

// out[dst[e],:] += al[e] * Hs[src[e],:]
__global__ void dn_scatter_node_feat(const float* __restrict__ Hs, const float* __restrict__ al,
                                     const int* __restrict__ src, const int* __restrict__ dst,
                                     float* __restrict__ out, int total) {
  int i = blockIdx.x * blockDim.x + threadIdx.x;
  if (i >= total) return;
  int e = i >> 7, j = i & 127;
  atomicAdd(&out[(size_t)dst[e] * 128 + j], al[e] * Hs[(size_t)src[e] * 128 + j]);
}

// out[batch[n],:] += al[n] * Hs[n,:]
__global__ void dn_scatter_mol(const float* __restrict__ Hs, const float* __restrict__ al,
                               const int* __restrict__ batch, float* __restrict__ out, int total) {
  int i = blockIdx.x * blockDim.x + threadIdx.x;
  if (i >= total) return;
  int n = i >> 7, j = i & 127;
  atomicAdd(&out[(size_t)batch[n] * 128 + j], al[n] * Hs[i]);
}

// out[idx[n],:] += X[n,:]
__global__ void dn_scatter_rows(const float* __restrict__ X, const int* __restrict__ idx,
                                float* __restrict__ out, int total) {
  int i = blockIdx.x * blockDim.x + threadIdx.x;
  if (i >= total) return;
  int n = i >> 7, j = i & 127;
  atomicAdd(&out[(size_t)idx[n] * 128 + j], X[i]);
}

__global__ void dn_bias_act(float* __restrict__ X, const float* __restrict__ bias,
                            int total, int dim, int act) {
  int i = blockIdx.x * blockDim.x + threadIdx.x;
  if (i >= total) return;
  int j = i % dim;
  float v = X[i] + (bias ? bias[j] : 0.f);
  X[i] = act_apply(v, act);
}

// GRU gates: gi/gh [rows,384] (biases already added in GEMM), hprev [rows,128]
__global__ void dn_gru_gate(const float* __restrict__ gi, const float* __restrict__ gh,
                            const float* __restrict__ hprev, float* __restrict__ hout,
                            int total, int dorelu) {
  int i = blockIdx.x * blockDim.x + threadIdx.x;
  if (i >= total) return;
  int n = i >> 7, j = i & 127;
  size_t b = (size_t)n * 384 + j;
  float r  = sigmoidf(gi[b] + gh[b]);
  float zz = sigmoidf(gi[b + 128] + gh[b + 128]);
  float nn = tanhf(gi[b + 256] + r * gh[b + 256]);
  float h  = (1.f - zz) * nn + zz * hprev[i];
  hout[i] = dorelu ? fmaxf(h, 0.f) : h;
}

// LSTM gates: gi/gh [rows,768], c [rows,192] in/out, hout [rows,192]
__global__ void dn_lstm_gate(const float* __restrict__ gi, const float* __restrict__ gh,
                             float* __restrict__ c, float* __restrict__ hout, int total) {
  int i = blockIdx.x * blockDim.x + threadIdx.x;
  if (i >= total) return;
  int n = i / 192, j = i % 192;
  size_t b = (size_t)n * 768 + j;
  float ig = sigmoidf(gi[b] + gh[b]);
  float fg = sigmoidf(gi[b + 192] + gh[b + 192]);
  float gg = tanhf(gi[b + 384] + gh[b + 384]);
  float og = sigmoidf(gi[b + 576] + gh[b + 576]);
  float cc = fg * c[i] + ig * gg;
  c[i] = cc;
  hout[i] = og * tanhf(cc);
}

// m[dst[e],:] += relu(xg[src[e],:] + e[e,:])
__global__ void dn_gine_msg(const float* __restrict__ xg, const float* __restrict__ ef,
                            const int* __restrict__ src, const int* __restrict__ dst,
                            float* __restrict__ m, int total) {
  int i = blockIdx.x * blockDim.x + threadIdx.x;
  if (i >= total) return;
  int e = i >> 7, j = i & 127;
  float v = xg[(size_t)src[e] * 128 + j] + ef[i];
  atomicAdd(&m[(size_t)dst[e] * 128 + j], fmaxf(v, 0.f));
}

__global__ void dn_gine_comb(const float* __restrict__ m, const float* __restrict__ xg,
                             const float* __restrict__ eps, float* __restrict__ t, int total) {
  int i = blockIdx.x * blockDim.x + threadIdx.x;
  if (i >= total) return;
  t[i] = m[i] + (1.f + eps[0]) * xg[i];
}

__global__ void dn_bn_relu(float* __restrict__ z, const float* __restrict__ g,
                           const float* __restrict__ b, int total, int dim, float scale) {
  int i = blockIdx.x * blockDim.x + threadIdx.x;
  if (i >= total) return;
  int j = i % dim;
  z[i] = fmaxf(z[i] * (g[j] * scale) + b[j], 0.f);
}

// JK attention: alpha[n,t] = softmax_t( dot([HF_t;HB_t], attW) + b )
__global__ void dn_jk_score(const float* __restrict__ HF, const float* __restrict__ HB,
                            const float* __restrict__ attW, const float* __restrict__ attB,
                            float* __restrict__ alpha, int rows, int N) {
  int n = blockIdx.x * blockDim.x + threadIdx.x;
  if (n >= rows) return;
  float s[3];
#pragma unroll
  for (int t = 0; t < 3; ++t) {
    const float* hf = HF + ((size_t)t * N + n) * 192;
    const float* hb = HB + ((size_t)t * N + n) * 192;
    float acc = attB[0];
    for (int j = 0; j < 192; ++j) acc += hf[j] * attW[j] + hb[j] * attW[192 + j];
    s[t] = acc;
  }
  float m = fmaxf(s[0], fmaxf(s[1], s[2]));
  float e0 = __expf(s[0] - m), e1 = __expf(s[1] - m), e2 = __expf(s[2] - m);
  float inv = 1.f / (e0 + e1 + e2);
  alpha[(size_t)n * 3 + 0] = e0 * inv;
  alpha[(size_t)n * 3 + 1] = e1 * inv;
  alpha[(size_t)n * 3 + 2] = e2 * inv;
}

__global__ void dn_jk_mix(const float* __restrict__ XS, const float* __restrict__ alpha,
                          float* __restrict__ out, int total, int N) {
  int i = blockIdx.x * blockDim.x + threadIdx.x;
  if (i >= total) return;
  int n = i >> 7, j = i & 127;
  float a0 = alpha[(size_t)n * 3 + 0];
  float a1 = alpha[(size_t)n * 3 + 1];
  float a2 = alpha[(size_t)n * 3 + 2];
  size_t st = (size_t)N * 128;
  out[i] = a0 * XS[i] + a1 * XS[st + i] + a2 * XS[2 * st + i];
}

__global__ void dn_count(const int* __restrict__ batch, float* __restrict__ cnt, int N) {
  int n = blockIdx.x * blockDim.x + threadIdx.x;
  if (n >= N) return;
  atomicAdd(&cnt[batch[n]], 1.f);
}

__global__ void dn_div_cnt(float* __restrict__ g2, const float* __restrict__ cnt, int total) {
  int i = blockIdx.x * blockDim.x + threadIdx.x;
  if (i >= total) return;
  g2[i] /= fmaxf(cnt[i >> 7], 1.f);
}

__global__ void dn_concat4(const float* __restrict__ a, const float* __restrict__ b,
                           const float* __restrict__ c, const float* __restrict__ d,
                           float* __restrict__ out, int total) {
  int i = blockIdx.x * blockDim.x + threadIdx.x;
  if (i >= total) return;
  int g = i >> 9, j = i & 511;
  float v;
  if (j < 128)      v = a[(size_t)g * 128 + j];
  else if (j < 256) v = b[(size_t)g * 128 + (j - 128)];
  else if (j < 384) v = c[(size_t)g * 128 + (j - 256)];
  else              v = d[(size_t)g * 128 + (j - 384)];
  out[i] = v;
}

// =============================== host driver ===============================
// Input slot indices assume JAX-pytree flattening (dict keys sorted) of the
// setup_inputs() dict: batch, edge_attr, edge_index, params{...}, x.
enum {
  IN_BATCH = 0, IN_EATTR = 1, IN_EIDX = 2,
  A_GATE_W1 = 3, A_GATE_W2, A_GATE_ATT_L, A_GATE_ATT_R, A_GATE_BIAS,
  A_GAT0_W, A_GAT0_ATT_DST, A_GAT0_ATT_SRC, A_GAT0_BIAS,
  A_GAT1_W, A_GAT1_ATT_DST, A_GAT1_ATT_SRC, A_GAT1_BIAS,
  A_GRU0_WH, A_GRU0_WI, A_GRU0_BH, A_GRU0_BI,
  A_GRUS0_WH, A_GRUS0_WI, A_GRUS0_BH, A_GRUS0_BI,
  A_GRUS1_WH, A_GRUS1_WI, A_GRUS1_BH, A_GRUS1_BI,
  A_LIN1_W, A_LIN1_B, A_LIN2_W, A_LIN2_B,
  A_MGAT_W, A_MGAT_ATT_DST, A_MGAT_ATT_SRC, A_MGAT_BIAS,
  A_MGRU_WH, A_MGRU_WI, A_MGRU_BH, A_MGRU_BI,                 // ..39
  FC1_BN_B, FC1_BN_G, FC1_L1_W, FC1_L1_B, FC1_L2_W, FC1_L2_B, // 40..45
  FC2_BN_B, FC2_BN_G, FC2_L1_W, FC2_L1_B, FC2_L2_W, FC2_L2_B, // 46..51
  FC3_L1_W, FC3_L1_B, FC3_L2_W, FC3_L2_B,
  FC3_L3_W, FC3_L3_B, FC3_L4_W, FC3_L4_B,                     // 52..59
  GC_BASE = 60,  // per conv (x3): eps, linE.W, linE.b, bn_b, bn_g, l1.W, l1.b, l2.W, l2.b
  JK_ATT_W = 87, JK_ATT_B,
  JK_BWD_WH = 89, JK_BWD_WI, JK_BWD_BH, JK_BWD_BI,
  JK_FWD_WH = 93, JK_FWD_WI, JK_FWD_BH, JK_FWD_BI,
  G_LIN_W = 97, G_LIN_B,
  IN_X = 99
};

static const float BN_SCALE_H = 0.9999950000374997f; // 1/sqrt(1+1e-5)

extern "C" void kernel_launch(void* const* d_in, const int* in_sizes, int n_in,
                              void* d_out, int out_size, void* d_ws, size_t ws_size,
                              hipStream_t stream) {
  (void)n_in; (void)out_size; (void)ws_size;

  auto F = [&](int i) { return (const float*)d_in[i]; };
  const int* batch = (const int*)d_in[IN_BATCH];
  const int* ei    = (const int*)d_in[IN_EIDX];
  const float* ea  = F(IN_EATTR);
  const float* x   = F(IN_X);

  const int N = in_sizes[IN_X] / 128;
  const int E = in_sizes[IN_EATTR] / 16;
  const int G = 1024;
  const int* src = ei;
  const int* dst = ei + E;

  // ---- workspace bump allocator ----
  uintptr_t wp = (uintptr_t)d_ws;
  auto alloc = [&](size_t bytes) -> float* {
    float* p = (float*)wp;
    wp += (bytes + 255) & ~(size_t)255;
    return p;
  };
  const size_t N128 = (size_t)N * 128 * 4, G128 = (size_t)G * 128 * 4;

  float* ea_n   = alloc((size_t)E * 16 * 4);
  float* zbuf   = alloc((size_t)E * 128 * 4);    // edge features (gate z / GINE e)
  float* h0     = alloc(N128);
  float* xs1    = alloc(N128);
  float* xa     = alloc(N128);
  float* hs     = alloc(N128);
  float* outs   = alloc(N128);                   // scatter target / gat h
  float* a_e    = alloc((size_t)E * 4);
  float* e_e    = alloc((size_t)E * 4);
  float* al_e   = alloc((size_t)E * 4);
  unsigned* mu  = (unsigned*)alloc((size_t)N * 4);
  float* ssum   = alloc((size_t)N * 4);
  float* xr     = alloc((size_t)N * 4);
  float* asn    = alloc((size_t)N * 4);
  float* adn    = alloc((size_t)N * 4);
  float* gi     = alloc((size_t)N * 768 * 4);    // shared GRU/LSTM pre-activations
  float* gh     = alloc((size_t)N * 768 * 4);
  float* out_g  = alloc(G128);
  float* hd     = alloc(G128);
  float* agg    = alloc(G128);
  float* graph1 = alloc(G128);
  float* graph2 = alloc(G128);
  float* XS     = alloc(3 * N128);               // GINE layer outputs
  float* HF     = alloc((size_t)3 * N * 192 * 4);
  float* HB     = alloc((size_t)3 * N * 192 * 4);
  float* cl     = alloc((size_t)N * 192 * 4);
  float* hz     = alloc((size_t)N * 192 * 4);    // zero LSTM h0
  float* alpha  = alloc((size_t)N * 3 * 4);
  float* xgjk   = alloc(N128);
  float* nodeo  = alloc(N128);
  float* cnt    = alloc((size_t)G * 4);
  float* o1     = alloc(G128);
  float* o2     = alloc(G128);
  float* t256   = alloc((size_t)G * 256 * 4);
  float* cat    = alloc((size_t)G * 512 * 4);
  float* zz1    = alloc((size_t)G * 1024 * 4);
  float* zz2    = alloc((size_t)G * 1024 * 4);
  float* zz3    = alloc((size_t)G * 512 * 4);
  float* mbuf   = alloc(N128);
  float* tbuf   = alloc(N128);
  float* z1n    = alloc(N128);

  auto fill = [&](float* p, float v, long long n) {
    dn_fill<<<CEILDIV((int)n, 256), 256, 0, stream>>>(p, v, (int)n);
  };
  auto gemm = [&](const float* A, int lda, const float* W, int ldw, const float* bias,
                  float* C, int ldc, int M, int Nc, int K, int act) {
    dim3 g(CEILDIV(Nc, 64), CEILDIV(M, 128));
    dn_gemm_wmma<<<g, dim3(256), 0, stream>>>(A, lda, W, ldw, bias, C, ldc, M, Nc, K, act);
  };
  auto seg_softmax = [&](const float* a, const int* idx, int ne, int nseg) {
    fill((float*)mu, 0.f, nseg);
    dn_seg_max<<<CEILDIV(ne, 256), 256, 0, stream>>>(a, idx, mu, ne);
    dn_seg_exp<<<CEILDIV(ne, 256), 256, 0, stream>>>(a, idx, mu, e_e, ne);
    fill(ssum, 0.f, nseg);
    dn_seg_sum<<<CEILDIV(ne, 256), 256, 0, stream>>>(e_e, idx, ssum, ne);
    dn_seg_alpha<<<CEILDIV(ne, 256), 256, 0, stream>>>(e_e, idx, ssum, al_e, ne);
  };
  const int EN = E * 128, NN = N * 128, GN = G * 128;

  // ==== edge attr row-normalize ====
  dn_rownorm16<<<CEILDIV(E, 256), 256, 0, stream>>>(ea, ea_n, E);

  // ==== AttentiveFP branch ====
  gemm(x, 128, F(A_LIN1_W), 128, F(A_LIN1_B), h0, 128, N, 128, 128, ACT_LEAKY);

  // gate conv: z = leaky([x_src, ea] W1^T) done as (h0 W1a^T)[src] + ea W1b^T
  gemm(h0, 128, F(A_GATE_W1), 144, nullptr, xs1, 128, N, 128, 128, ACT_NONE);
  gemm(ea_n, 16, F(A_GATE_W1) + 128, 144, nullptr, zbuf, 128, E, 128, 16, ACT_NONE);
  dn_edge_gate_z<<<CEILDIV(EN, 256), 256, 0, stream>>>(xs1, zbuf, src, EN);
  dn_dot<<<CEILDIV(N, 256), 256, 0, stream>>>(h0, F(A_GATE_ATT_R), xr, N);
  dn_edge_gate_score<<<CEILDIV(E, 256), 256, 0, stream>>>(zbuf, F(A_GATE_ATT_L), xr, dst, a_e, E);
  seg_softmax(a_e, dst, E, N);
  fill(outs, 0.f, NN);
  dn_scatter_edge_feat<<<CEILDIV(EN, 256), 256, 0, stream>>>(zbuf, al_e, dst, outs, EN);
  // h = elu(segsum(al*z) @ W2^T + bias)   (linearity of GEMM over the segment sum)
  gemm(outs, 128, F(A_GATE_W2), 128, F(A_GATE_BIAS), hs, 128, N, 128, 128, ACT_ELU);
  // GRU0: xa = relu(gru(h, h0))
  gemm(hs, 128, F(A_GRU0_WI), 128, F(A_GRU0_BI), gi, 384, N, 384, 128, ACT_NONE);
  gemm(h0, 128, F(A_GRU0_WH), 128, F(A_GRU0_BH), gh, 384, N, 384, 128, ACT_NONE);
  dn_gru_gate<<<CEILDIV(NN, 256), 256, 0, stream>>>(gi, gh, h0, xa, NN, 1);

  // 2 GAT+GRU layers
  const int gatW[2]  = {A_GAT0_W, A_GAT1_W};
  const int gatAS[2] = {A_GAT0_ATT_SRC, A_GAT1_ATT_SRC};
  const int gatAD[2] = {A_GAT0_ATT_DST, A_GAT1_ATT_DST};
  const int gatB[2]  = {A_GAT0_BIAS, A_GAT1_BIAS};
  const int gruWI[2] = {A_GRUS0_WI, A_GRUS1_WI};
  const int gruWH[2] = {A_GRUS0_WH, A_GRUS1_WH};
  const int gruBI[2] = {A_GRUS0_BI, A_GRUS1_BI};
  const int gruBH[2] = {A_GRUS0_BH, A_GRUS1_BH};
  for (int l = 0; l < 2; ++l) {
    gemm(xa, 128, F(gatW[l]), 128, nullptr, hs, 128, N, 128, 128, ACT_NONE);
    dn_dot2<<<CEILDIV(N, 256), 256, 0, stream>>>(hs, F(gatAS[l]), F(gatAD[l]), asn, adn, N);
    dn_edge_score<<<CEILDIV(E, 256), 256, 0, stream>>>(asn, adn, src, dst, a_e, E);
    seg_softmax(a_e, dst, E, N);
    fill(outs, 0.f, NN);
    dn_scatter_node_feat<<<CEILDIV(EN, 256), 256, 0, stream>>>(hs, al_e, src, dst, outs, EN);
    dn_bias_act<<<CEILDIV(NN, 256), 256, 0, stream>>>(outs, F(gatB[l]), NN, 128, ACT_ELU);
    gemm(outs, 128, F(gruWI[l]), 128, F(gruBI[l]), gi, 384, N, 384, 128, ACT_NONE);
    gemm(xa, 128, F(gruWH[l]), 128, F(gruBH[l]), gh, 384, N, 384, 128, ACT_NONE);
    dn_gru_gate<<<CEILDIV(NN, 256), 256, 0, stream>>>(gi, gh, xa, xa, NN, 1);
  }

  // graph pooling + 3 molecule-level GAT/GRU steps
  fill(out_g, 0.f, GN);
  dn_scatter_rows<<<CEILDIV(NN, 256), 256, 0, stream>>>(xa, batch, out_g, NN);
  dn_bias_act<<<CEILDIV(GN, 256), 256, 0, stream>>>(out_g, nullptr, GN, 128, ACT_RELU);
  gemm(xa, 128, F(A_MGAT_W), 128, nullptr, hs, 128, N, 128, 128, ACT_NONE);
  dn_dot<<<CEILDIV(N, 256), 256, 0, stream>>>(hs, F(A_MGAT_ATT_SRC), asn, N);
  for (int t = 0; t < 3; ++t) {
    gemm(out_g, 128, F(A_MGAT_W), 128, nullptr, hd, 128, G, 128, 128, ACT_NONE);
    dn_dot<<<CEILDIV(G, 256), 256, 0, stream>>>(hd, F(A_MGAT_ATT_DST), adn, G);
    dn_mol_score<<<CEILDIV(N, 256), 256, 0, stream>>>(asn, adn, batch, a_e, N);
    seg_softmax(a_e, batch, N, G);
    fill(agg, 0.f, GN);
    dn_scatter_mol<<<CEILDIV(NN, 256), 256, 0, stream>>>(hs, al_e, batch, agg, NN);
    dn_bias_act<<<CEILDIV(GN, 256), 256, 0, stream>>>(agg, F(A_MGAT_BIAS), GN, 128, ACT_ELU);
    gemm(agg, 128, F(A_MGRU_WI), 128, F(A_MGRU_BI), gi, 384, G, 384, 128, ACT_NONE);
    gemm(out_g, 128, F(A_MGRU_WH), 128, F(A_MGRU_BH), gh, 384, G, 384, 128, ACT_NONE);
    dn_gru_gate<<<CEILDIV(GN, 256), 256, 0, stream>>>(gi, gh, out_g, out_g, GN, 1);
  }
  gemm(out_g, 128, F(A_LIN2_W), 128, F(A_LIN2_B), graph1, 128, G, 128, 128, ACT_NONE);

  // ==== GINE branch ====
  const float* xg = x;
  for (int l = 0; l < 3; ++l) {
    const int b0 = GC_BASE + 9 * l; // eps, linE.W, linE.b, bn_b, bn_g, l1.W, l1.b, l2.W, l2.b
    gemm(ea_n, 16, F(b0 + 1), 16, F(b0 + 2), zbuf, 128, E, 128, 16, ACT_NONE);
    fill(mbuf, 0.f, NN);
    dn_gine_msg<<<CEILDIV(EN, 256), 256, 0, stream>>>(xg, zbuf, src, dst, mbuf, EN);
    dn_gine_comb<<<CEILDIV(NN, 256), 256, 0, stream>>>(mbuf, xg, F(b0 + 0), tbuf, NN);
    gemm(tbuf, 128, F(b0 + 5), 128, F(b0 + 6), z1n, 128, N, 128, 128, ACT_NONE);
    dn_bn_relu<<<CEILDIV(NN, 256), 256, 0, stream>>>(z1n, F(b0 + 4), F(b0 + 3), NN, 128, BN_SCALE_H);
    float* xs_l = XS + (size_t)l * N * 128;
    gemm(z1n, 128, F(b0 + 7), 128, F(b0 + 8), xs_l, 128, N, 128, 128, ACT_RELU);
    xg = xs_l;
  }

  // JK bi-LSTM over the 3 layer outputs (hidden 192 per direction)
  const long long NL = (long long)N * 192;
  fill(hz, 0.f, NL); fill(cl, 0.f, NL);
  const float* hprev = hz;
  for (int t = 0; t < 3; ++t) { // forward
    gemm(XS + (size_t)t * N * 128, 128, F(JK_FWD_WI), 128, F(JK_FWD_BI), gi, 768, N, 768, 128, ACT_NONE);
    gemm(hprev, 192, F(JK_FWD_WH), 192, F(JK_FWD_BH), gh, 768, N, 768, 192, ACT_NONE);
    float* hslot = HF + (size_t)t * N * 192;
    dn_lstm_gate<<<CEILDIV((int)NL, 256), 256, 0, stream>>>(gi, gh, cl, hslot, (int)NL);
    hprev = hslot;
  }
  fill(cl, 0.f, NL);
  hprev = hz;
  for (int s = 0; s < 3; ++s) { // backward (consumes layers in reverse)
    const int t = 2 - s;
    gemm(XS + (size_t)t * N * 128, 128, F(JK_BWD_WI), 128, F(JK_BWD_BI), gi, 768, N, 768, 128, ACT_NONE);
    gemm(hprev, 192, F(JK_BWD_WH), 192, F(JK_BWD_BH), gh, 768, N, 768, 192, ACT_NONE);
    float* hslot = HB + (size_t)t * N * 192;
    dn_lstm_gate<<<CEILDIV((int)NL, 256), 256, 0, stream>>>(gi, gh, cl, hslot, (int)NL);
    hprev = hslot;
  }
  dn_jk_score<<<CEILDIV(N, 256), 256, 0, stream>>>(HF, HB, F(JK_ATT_W), F(JK_ATT_B), alpha, N, N);
  dn_jk_mix<<<CEILDIV(NN, 256), 256, 0, stream>>>(XS, alpha, xgjk, NN, N);
  gemm(xgjk, 128, F(G_LIN_W), 128, F(G_LIN_B), nodeo, 128, N, 128, 128, ACT_NONE);
  fill(graph2, 0.f, GN);
  dn_scatter_rows<<<CEILDIV(NN, 256), 256, 0, stream>>>(nodeo, batch, graph2, NN);
  fill(cnt, 0.f, G);
  dn_count<<<CEILDIV(N, 256), 256, 0, stream>>>(batch, cnt, N);
  dn_div_cnt<<<CEILDIV(GN, 256), 256, 0, stream>>>(graph2, cnt, GN);

  // ==== heads ====
  gemm(graph1, 128, F(FC1_L1_W), 128, F(FC1_L1_B), t256, 256, G, 256, 128, ACT_NONE);
  dn_bn_relu<<<CEILDIV(G * 256, 256), 256, 0, stream>>>(t256, F(FC1_BN_G), F(FC1_BN_B), G * 256, 256, BN_SCALE_H);
  gemm(t256, 256, F(FC1_L2_W), 256, F(FC1_L2_B), o1, 128, G, 128, 256, ACT_NONE);
  gemm(graph2, 128, F(FC2_L1_W), 128, F(FC2_L1_B), t256, 256, G, 256, 128, ACT_NONE);
  dn_bn_relu<<<CEILDIV(G * 256, 256), 256, 0, stream>>>(t256, F(FC2_BN_G), F(FC2_BN_B), G * 256, 256, BN_SCALE_H);
  gemm(t256, 256, F(FC2_L2_W), 256, F(FC2_L2_B), o2, 128, G, 128, 256, ACT_NONE);

  dn_concat4<<<CEILDIV(G * 512, 256), 256, 0, stream>>>(graph1, o1, graph2, o2, cat, G * 512);
  gemm(cat, 512, F(FC3_L1_W), 512, F(FC3_L1_B), zz1, 1024, G, 1024, 512, ACT_RELU);
  gemm(zz1, 1024, F(FC3_L2_W), 1024, F(FC3_L2_B), zz2, 1024, G, 1024, 1024, ACT_RELU);
  gemm(zz2, 1024, F(FC3_L3_W), 1024, F(FC3_L3_B), zz3, 512, G, 512, 1024, ACT_RELU);
  gemm(zz3, 512, F(FC3_L4_W), 512, F(FC3_L4_B), (float*)d_out, 1, G, 1, 512, ACT_NONE);
}
